// InteractionDynamics_16870631539082
// MI455X (gfx1250) — matrix operand order repaired and checked by reading, not compiled, not run on hardware
//
#include <hip/hip_runtime.h>
#include <hip/hip_bf16.h>

// ---------------------------------------------------------------------------
// InteractionDynamics on MI455X (gfx1250): bf16 WMMA pairwise MLP.
// B=8, T=24, N=96, D=32, H=512.  Output: (B, T-1=23, N, D) fp32.
// ---------------------------------------------------------------------------

typedef __attribute__((ext_vector_type(16))) __bf16 v16bf;
typedef __attribute__((ext_vector_type(8)))  float  v8f;
typedef __attribute__((ext_vector_type(4)))  unsigned int u32x4;

#define BB 8
#define TT 24
#define NN 96
#define DD 32
#define HH 512
#define TO 23              // T-1 output steps
#define BND (BB*NN*DD)     // 24576 floats per state slot

union Frag16 { v16bf v; u32x4 q[2]; };

// Load B fragment: 8 bf16 at p, 8 bf16 at p+16 (col n, K = half*8.. and 16+half*8..)
__device__ __forceinline__ v16bf load_bfrag(const __bf16* p) {
    Frag16 f;
    f.q[0] = *(const u32x4*)(p);
    f.q[1] = *(const u32x4*)(p + 16);
    return f.v;
}
// Load A fragment: 8 bf16 at p, 8 bf16 at p+16 (K = half*8.. and 16+half*8..)
__device__ __forceinline__ v16bf load_afrag(const __bf16* p) {
    Frag16 f;
    f.q[0] = *(const u32x4*)(p);
    f.q[1] = *(const u32x4*)(p + 16);
    return f.v;
}

// ---------------------------------------------------------------------------
// Init / prep kernels
// ---------------------------------------------------------------------------
__global__ void k_zero(float* p, int n) {
    int i = blockIdx.x * 256 + threadIdx.x;
    if (i < n) p[i] = 0.f;
}

// Transpose+convert W1,W2,W3 (512x512 fp32, (K,N) row-major) -> (N,K) bf16.
__global__ void k_transpose3(const float* __restrict__ W1,
                             const float* __restrict__ W2,
                             const float* __restrict__ W3,
                             __bf16* __restrict__ T1,
                             __bf16* __restrict__ T2,
                             __bf16* __restrict__ T3) {
    int idx = blockIdx.x * 256 + threadIdx.x;   // 0 .. 3*512*512-1
    int sel = idx >> 18;
    int r   = idx & 262143;
    int n = r >> 9, k = r & 511;
    const float* W = (sel == 0) ? W1 : (sel == 1) ? W2 : W3;
    __bf16*      T = (sel == 0) ? T1 : (sel == 1) ? T2 : T3;
    T[n * 512 + k] = (__bf16)W[k * 512 + n];
}

// ---------------------------------------------------------------------------
// State recurrence: s_t = exist*x_t + (1-exist)*s_{t-1}.
// S holds slots: S[0..1]=zeros, S[t+2] = s_t.
// ---------------------------------------------------------------------------
__global__ void k_state(const float* __restrict__ x, float* __restrict__ S, int t) {
    int idx = blockIdx.x * 256 + threadIdx.x;      // (b,n,d) flat, < 24576
    int d = idx & 31;
    int n = (idx >> 5) % NN;
    int b = idx / (NN * DD);
    const float* xt = x + ((size_t)(b * TT + t) * NN + n) * DD;
    float exist = xt[1];
    float prev  = S[(t + 1) * BND + idx];
    S[(t + 2) * BND + idx] = exist * xt[d] + (1.f - exist) * prev;
}

// ---------------------------------------------------------------------------
// Encoder factorization:
//   scat = [s_{t-2} | s_{t-1} | s_t]  (96 per (b,n))
//   U = scat @ W_enc[0:96,:]   V = scat @ W_enc[96:192,:]
// ---------------------------------------------------------------------------
__global__ void k_encoder(const float* __restrict__ S,
                          const float* __restrict__ W_enc,
                          float* __restrict__ U, float* __restrict__ V, int t) {
    __shared__ float scat[96];
    int bn  = blockIdx.x;          // b*96+n
    int tid = threadIdx.x;
    if (tid < 96) {
        int slot = tid >> 5, d = tid & 31;
        scat[tid] = S[(t + slot) * BND + bn * DD + d];
    }
    __syncthreads();
    for (int c = tid; c < HH; c += 256) {
        float su = 0.f, sv = 0.f;
        #pragma unroll 4
        for (int k = 0; k < 96; ++k) {
            float s = scat[k];
            su += s * W_enc[k * HH + c];
            sv += s * W_enc[(96 + k) * HH + c];
        }
        U[(size_t)bn * HH + c] = su;
        V[(size_t)bn * HH + c] = sv;
    }
}

// ---------------------------------------------------------------------------
// Pairwise WMMA layers. One workgroup = one (b,i); rows = all 96 j's.
// Register blocking: each wave owns N-tiles {wave, wave+16} simultaneously
// (12 accumulators) so one A-fragment LDS load feeds two WMMAs, and the two
// B-fragments are double-buffered in registers to hide L2 latency.
// ---------------------------------------------------------------------------
__device__ __forceinline__ void layer_ff(const __bf16* __restrict__ hin,
                                         __bf16* __restrict__ hout,
                                         const __bf16* __restrict__ Wt,
                                         const float* __restrict__ bias,
                                         int wave, int lane) {
    const int l15  = lane & 15;
    const int half = lane >> 4;
    const int ncol0 = wave * 16 + l15;           // N-tile: wave
    const int ncol1 = ncol0 + 256;               // N-tile: wave+16
    const __bf16* w0 = Wt + (size_t)ncol0 * HH + half * 8;
    const __bf16* w1 = Wt + (size_t)ncol1 * HH + half * 8;

    v8f acc[12];
    #pragma unroll
    for (int i = 0; i < 12; ++i) acc[i] = (v8f){0,0,0,0,0,0,0,0};

    v16bf b0 = load_bfrag(w0);
    v16bf b1 = load_bfrag(w1);
    for (int kk = 0; kk < HH; kk += 32) {
        // Prefetch next k-step's B fragments (overread at kk==480 stays in ws)
        v16bf nb0 = load_bfrag(w0 + kk + 32);
        v16bf nb1 = load_bfrag(w1 + kk + 32);
        #pragma unroll
        for (int mt = 0; mt < 6; ++mt) {
            v16bf a = load_afrag(hin + (mt * 16 + l15) * HH + kk + half * 8);
            acc[mt]     = __builtin_amdgcn_wmma_f32_16x16x32_bf16(
                false, a, false, b0, (short)0, acc[mt],     false, false);
            acc[6 + mt] = __builtin_amdgcn_wmma_f32_16x16x32_bf16(
                false, a, false, b1, (short)0, acc[6 + mt], false, false);
        }
        b0 = nb0; b1 = nb1;
    }

    const float bn0 = bias[ncol0];
    const float bn1 = bias[ncol1];
    #pragma unroll
    for (int mt = 0; mt < 6; ++mt) {
        const int rbase = mt * 16 + half * 8;
        #pragma unroll
        for (int r = 0; r < 8; ++r) {
            hout[(rbase + r) * HH + ncol0] = (__bf16)fmaxf(acc[mt][r]     + bn0, 0.f);
            hout[(rbase + r) * HH + ncol1] = (__bf16)fmaxf(acc[6 + mt][r] + bn1, 0.f);
        }
    }
}

// Final layer: h @ W3 + b3, column-summed over all 96 rows into eacc.
__device__ __forceinline__ void layer_reduce(const __bf16* __restrict__ hin,
                                             const __bf16* __restrict__ Wt,
                                             const float* __restrict__ bias,
                                             float* __restrict__ eacc,
                                             int wave, int lane) {
    const int l15  = lane & 15;
    const int half = lane >> 4;
    const int ncol0 = wave * 16 + l15;
    const int ncol1 = ncol0 + 256;
    const __bf16* w0 = Wt + (size_t)ncol0 * HH + half * 8;
    const __bf16* w1 = Wt + (size_t)ncol1 * HH + half * 8;

    v8f acc[12];
    #pragma unroll
    for (int i = 0; i < 12; ++i) acc[i] = (v8f){0,0,0,0,0,0,0,0};

    v16bf b0 = load_bfrag(w0);
    v16bf b1 = load_bfrag(w1);
    for (int kk = 0; kk < HH; kk += 32) {
        v16bf nb0 = load_bfrag(w0 + kk + 32);
        v16bf nb1 = load_bfrag(w1 + kk + 32);
        #pragma unroll
        for (int mt = 0; mt < 6; ++mt) {
            v16bf a = load_afrag(hin + (mt * 16 + l15) * HH + kk + half * 8);
            acc[mt]     = __builtin_amdgcn_wmma_f32_16x16x32_bf16(
                false, a, false, b0, (short)0, acc[mt],     false, false);
            acc[6 + mt] = __builtin_amdgcn_wmma_f32_16x16x32_bf16(
                false, a, false, b1, (short)0, acc[6 + mt], false, false);
        }
        b0 = nb0; b1 = nb1;
    }

    const float bn0 = bias[ncol0];
    const float bn1 = bias[ncol1];
    #pragma unroll
    for (int mt = 0; mt < 6; ++mt) {
        float s0 = 8.f * bn0;                      // bias for 8 rows
        float s1 = 8.f * bn1;
        #pragma unroll
        for (int r = 0; r < 8; ++r) { s0 += acc[mt][r]; s1 += acc[6 + mt][r]; }
        s0 += __shfl_xor(s0, 16, 32);              // combine row halves
        s1 += __shfl_xor(s1, 16, 32);
        if (half == 0) {
            atomicAdd(&eacc[ncol0], s0);           // ds_add_f32
            atomicAdd(&eacc[ncol1], s1);
        }
    }
}

#define SMEM_BYTES (98304 * 2 + 2048)

__global__ __launch_bounds__(512) void
k_pairwise(const float* __restrict__ U, const float* __restrict__ V,
           const float* __restrict__ b_enc,
           const __bf16* __restrict__ WT1, const float* __restrict__ b1,
           const __bf16* __restrict__ WT2, const float* __restrict__ b2,
           const __bf16* __restrict__ WT3, const float* __restrict__ b3,
           float* __restrict__ EMB) {
    extern __shared__ __align__(16) char smem[];
    __bf16* hA   = (__bf16*)smem;                 // 96 x 512 bf16
    __bf16* hB   = (__bf16*)(smem + 98304);       // 96 x 512 bf16
    float*  eacc = (float*)(smem + 196608);       // 512 f32

    const int tid  = threadIdx.x;
    const int wave = tid >> 5;
    const int lane = tid & 31;
    const int bi   = blockIdx.x;                  // b*96 + i
    const int b    = bi / NN;

    // h0[j,c] = relu(u_i[c] + v_j[c] + b_enc[c])
    const float* urow = U + (size_t)bi * HH;
    for (int idx = tid; idx < NN * HH; idx += 512) {
        int j = idx >> 9, c = idx & 511;
        float v = urow[c] + V[((size_t)b * NN + j) * HH + c] + b_enc[c];
        hA[idx] = (__bf16)fmaxf(v, 0.f);
    }
    eacc[tid] = 0.f;
    __syncthreads();

    layer_ff(hA, hB, WT1, b1, wave, lane);   __syncthreads();
    layer_ff(hB, hA, WT2, b2, wave, lane);   __syncthreads();
    layer_reduce(hA, WT3, b3, eacc, wave, lane);
    __syncthreads();

    EMB[(size_t)bi * HH + tid] = eacc[tid] * (1.f / 96.f);
}

// ---------------------------------------------------------------------------
// Decoder: out = relu(emb @ Wd1 + bd1) @ Wd2 + bd2   (tiny, VALU fp32)
// ---------------------------------------------------------------------------
__global__ void k_decoder(const float* __restrict__ EMB,
                          const float* __restrict__ Wd1, const float* __restrict__ bd1,
                          const float* __restrict__ Wd2, const float* __restrict__ bd2,
                          float* __restrict__ out, int t) {
    __shared__ float erow[HH];
    __shared__ float hd[HH];
    int bn  = blockIdx.x;            // b*96+n
    int tid = threadIdx.x;
    for (int c = tid; c < HH; c += 256) erow[c] = EMB[(size_t)bn * HH + c];
    __syncthreads();
    for (int c = tid; c < HH; c += 256) {
        float s = bd1[c];
        #pragma unroll 4
        for (int k = 0; k < HH; ++k) s += erow[k] * Wd1[k * HH + c];
        hd[c] = fmaxf(s, 0.f);
    }
    __syncthreads();
    if (tid < DD) {
        float s = bd2[tid];
        #pragma unroll 4
        for (int k = 0; k < HH; ++k) s += hd[k] * Wd2[k * DD + tid];
        int b = bn / NN, n = bn % NN;
        out[(((size_t)b * TO + t) * NN + n) * DD + tid] = s;
    }
}

// ---------------------------------------------------------------------------
// Host launcher
// ---------------------------------------------------------------------------
extern "C" void kernel_launch(void* const* d_in, const int* in_sizes, int n_in,
                              void* d_out, int out_size, void* d_ws, size_t ws_size,
                              hipStream_t stream) {
    const float* x     = (const float*)d_in[0];
    const float* W_enc = (const float*)d_in[1];
    const float* b_enc = (const float*)d_in[2];
    const float* W1    = (const float*)d_in[3];
    const float* b1    = (const float*)d_in[4];
    const float* W2    = (const float*)d_in[5];
    const float* b2    = (const float*)d_in[6];
    const float* W3    = (const float*)d_in[7];
    const float* b3    = (const float*)d_in[8];
    const float* Wd1   = (const float*)d_in[9];
    const float* bd1   = (const float*)d_in[10];
    const float* Wd2   = (const float*)d_in[11];
    const float* bd2   = (const float*)d_in[12];
    float* out = (float*)d_out;

    char* ws = (char*)d_ws;
    __bf16* WT1 = (__bf16*)(ws + 0);          // 512*512 bf16
    __bf16* WT2 = (__bf16*)(ws + 524288);
    __bf16* WT3 = (__bf16*)(ws + 1048576);
    float*  S   = (float*)(ws + 1572864);     // 26 slots * 24576 f32
    float*  U   = (float*)(ws + 4128768);     // 768*512 f32
    float*  V   = (float*)(ws + 5701632);
    float*  EMB = (float*)(ws + 7274496);

    (void)in_sizes; (void)n_in; (void)out_size; (void)ws_size;

    hipFuncSetAttribute((const void*)k_pairwise,
                        hipFuncAttributeMaxDynamicSharedMemorySize, SMEM_BYTES);

    // zero S slots 0 and 1 (initial carry zeros)
    k_zero<<<dim3((2 * BND + 255) / 256), dim3(256), 0, stream>>>(S, 2 * BND);
    // transpose + bf16-convert the three HxH weights
    k_transpose3<<<dim3(3072), dim3(256), 0, stream>>>(W1, W2, W3, WT1, WT2, WT3);

    for (int t = 0; t < TO; ++t) {
        k_state<<<dim3(BND / 256), dim3(256), 0, stream>>>(x, S, t);
        k_encoder<<<dim3(BB * NN), dim3(256), 0, stream>>>(S, W_enc, U, V, t);
        k_pairwise<<<dim3(BB * NN), dim3(512), SMEM_BYTES, stream>>>(
            U, V, b_enc, WT1, b1, WT2, b2, WT3, b3, EMB);
        k_decoder<<<dim3(BB * NN), dim3(256), 0, stream>>>(EMB, Wd1, bd1, Wd2, bd2, out, t);
    }
}
